// HybridQNN_9895604650015
// MI455X (gfx1250) — matrix operand order repaired and checked by reading, not compiled, not run on hardware
//
#include <hip/hip_runtime.h>
#include <hip/hip_bf16.h>
#include <math.h>

#define NQ   7
#define DIM  128      // 2^NQ statevector size
#define KTOT 256      // stacked [psi_r | psi_i]
#define D_IN 104
#define SAMPB 64      // samples per block in main kernel

typedef __attribute__((ext_vector_type(16))) _Float16 v16h;
typedef __attribute__((ext_vector_type(8)))  _Float16 v8h;
typedef __attribute__((ext_vector_type(8)))  float    v8f;
typedef __attribute__((ext_vector_type(4)))  float    v4f;

union AFrag { v16h v; _Float16 e[16]; };
union CFrag { v8f  v; float    e[8];  };
union HFrag { v8h  v; _Float16 e[8];  };

// ---------------------------------------------------------------------------
// Kernel 1: build the fixed 128x128 complex unitary of the entangling layers.
// Block j simulates basis column e_j through 3 layers of Rot + CNOT ring.
// Emits Br/Bi in WMMA-B-friendly row-major [n][k] f16, k = stacked re/im.
//   Br[n][k]      = Ur[n][k]     Br[n][128+k] = -Ui[n][k]
//   Bi[n][k]      = Ui[n][k]     Bi[n][128+k] =  Ur[n][k]
// ---------------------------------------------------------------------------
__global__ __launch_bounds__(DIM) void build_unitary_kernel(
    const float* __restrict__ qw, _Float16* __restrict__ Br,
    _Float16* __restrict__ Bi)
{
  __shared__ float b0r[DIM], b0i[DIM], b1r[DIM], b1i[DIM];
  const int j = blockIdx.x;
  const int k = threadIdx.x;
  float *cr = b0r, *ci = b0i, *nr = b1r, *ni = b1i;
  cr[k] = (k == j) ? 1.0f : 0.0f;
  ci[k] = 0.0f;
  __syncthreads();

  for (int l = 0; l < 3; ++l) {
    // Rot(phi, theta, omega) on each wire
    for (int w = 0; w < NQ; ++w) {
      const float phi = qw[(l * NQ + w) * 3 + 0];
      const float the = qw[(l * NQ + w) * 3 + 1];
      const float ome = qw[(l * NQ + w) * 3 + 2];
      float st, ct; sincosf(0.5f * the, &st, &ct);
      float sp, cp; sincosf(-0.5f * (phi + ome), &sp, &cp);   // m00 phase
      float sm, cm; sincosf(0.5f * (phi - ome), &sm, &cm);    // m01/m10 phase
      const float m00r =  cp * ct, m00i =  sp * ct;
      const float m01r = -cm * st, m01i = -sm * st;
      const float m10r =  cm * st, m10i = -sm * st;
      const float m11r =  cp * ct, m11i = -sp * ct;
      const int mask = 1 << (NQ - 1 - w);
      const int p = k ^ mask;
      float outr, outi;
      if ((k & mask) == 0) {
        const float r0 = cr[k], i0 = ci[k], r1 = cr[p], i1 = ci[p];
        outr = m00r * r0 - m00i * i0 + m01r * r1 - m01i * i1;
        outi = m00r * i0 + m00i * r0 + m01r * i1 + m01i * r1;
      } else {
        const float r0 = cr[p], i0 = ci[p], r1 = cr[k], i1 = ci[k];
        outr = m10r * r0 - m10i * i0 + m11r * r1 - m11i * i1;
        outi = m10r * i0 + m10i * r0 + m11r * i1 + m11i * r1;
      }
      nr[k] = outr; ni[k] = outi;
      __syncthreads();
      float* t = cr; cr = nr; nr = t; t = ci; ci = ni; ni = t;
    }
    // CNOT ring with range r
    const int rr = (l % (NQ - 1)) + 1;
    for (int w = 0; w < NQ; ++w) {
      const int cmask = 1 << (NQ - 1 - w);
      const int tmask = 1 << (NQ - 1 - ((w + rr) % NQ));
      const int src = (k & cmask) ? (k ^ tmask) : k;
      const float outr = cr[src];
      const float outi = ci[src];
      nr[k] = outr; ni[k] = outi;
      __syncthreads();
      float* t = cr; cr = nr; nr = t; t = ci; ci = ni; ni = t;
    }
  }

  const float re = cr[k];
  const float im = ci[k];
  Br[(size_t)k * KTOT + j]       = (_Float16)re;
  Br[(size_t)k * KTOT + DIM + j] = (_Float16)(-im);
  Bi[(size_t)k * KTOT + j]       = (_Float16)im;
  Bi[(size_t)k * KTOT + DIM + j] = (_Float16)re;
}

// ---------------------------------------------------------------------------
// Kernel 2: full pipeline per 64-sample block.
//   phase 0: stage x rows (b128, non-temporal) + pre_W into LDS
//   phase 1: h_w = tanh(x.W^T + b)*pi, cos/sin of h/2 (4 lanes per sample)
//   phase 2: product-state psi tile -> f16 LDS [64][256], 16B packed stores
//   phase 3: per-wave WMMA f32_16x16x32_f16 with double-buffered B frags,
//            then out = sum_k |phi_k|^2 * g_k + lin_b via shuffle reduction.
// ---------------------------------------------------------------------------
__global__ __launch_bounds__(256) void qnn_main_kernel(
    const float* __restrict__ x, const float* __restrict__ preW,
    const float* __restrict__ preb, const float* __restrict__ linW,
    const float* __restrict__ linb, const _Float16* __restrict__ Br,
    const _Float16* __restrict__ Bi, float* __restrict__ out)
{
  __shared__ __align__(16) char sbuf[SAMPB * KTOT * sizeof(_Float16)]; // 32KB
  __shared__ float sW[NQ * D_IN];
  __shared__ float scs[SAMPB][16];   // [s][0..6]=cos(h/2), [s][8..14]=sin(h/2)
  __shared__ float spart[8][16];
  float*    sx   = (float*)sbuf;        // [SAMPB][D_IN]  (phases 0-1)
  _Float16* sPsi = (_Float16*)sbuf;     // [SAMPB][KTOT]  (phases 2-3)

  const int tid  = threadIdx.x;
  const int base = blockIdx.x * SAMPB;

  // ---- phase 0: 16B-vector, non-temporal stage of x tile; pre_W to LDS ----
  {
    const v4f* xg  = (const v4f*)(x + (size_t)base * D_IN);  // rows are 26*16B
    v4f*       sx4 = (v4f*)sx;
    for (int i = tid; i < SAMPB * D_IN / 4; i += 256)
      sx4[i] = __builtin_nontemporal_load(&xg[i]);
    for (int i = tid; i < NQ * D_IN; i += 256)
      sW[i] = preW[i];
  }
  __syncthreads();

  // ---- phase 1: h = tanh(x.W^T + b)*pi; 4 lanes per sample ----
  {
    const int s = tid >> 2;
    const int part = tid & 3;
    float acc[NQ];
#pragma unroll
    for (int w = 0; w < NQ; ++w) acc[w] = 0.0f;
    const float* xr = sx + s * D_IN + part * 26;
    for (int jj = 0; jj < 26; ++jj) {
      const float xv = xr[jj];
      const int col = part * 26 + jj;
#pragma unroll
      for (int w = 0; w < NQ; ++w) acc[w] += xv * sW[w * D_IN + col];
    }
#pragma unroll
    for (int w = 0; w < NQ; ++w) {
      acc[w] += __shfl_xor(acc[w], 1, 32);
      acc[w] += __shfl_xor(acc[w], 2, 32);
    }
    if (part == 0) {
#pragma unroll
      for (int w = 0; w < NQ; ++w) {
        const float h = tanhf(acc[w] + preb[w]) * 3.14159265358979f;
        float sv, cv; sincosf(0.5f * h, &sv, &cv);
        scs[s][w] = cv; scs[s][w + 8] = sv;
      }
    }
  }
  __syncthreads();

  // ---- phase 2: product-state amplitudes (re|im stacked, f16, 16B stores) --
  {
    const int s = tid >> 2;
    const int part = tid & 3;
    float cw[NQ], swv[NQ];
#pragma unroll
    for (int w = 0; w < NQ; ++w) { cw[w] = scs[s][w]; swv[w] = scs[s][w + 8]; }
#pragma unroll
    for (int blk = 0; blk < 4; ++blk) {
      const int k0 = part * 32 + blk * 8;
      HFrag vr, vi;
#pragma unroll
      for (int u = 0; u < 8; ++u) {
        const int k = k0 + u;
        float mag = 1.0f;
#pragma unroll
        for (int w = 0; w < NQ; ++w)
          mag *= ((k >> (NQ - 1 - w)) & 1) ? swv[w] : cw[w];
        const int pc = __popc(k) & 3;              // (-i)^popcount phase
        float pr = 0.0f, pi_ = 0.0f;
        if      (pc == 0) pr  =  mag;
        else if (pc == 1) pi_ = -mag;
        else if (pc == 2) pr  = -mag;
        else              pi_ =  mag;
        vr.e[u] = (_Float16)pr;
        vi.e[u] = (_Float16)pi_;
      }
      *(v8h*)(sPsi + s * KTOT + k0)       = vr.v;
      *(v8h*)(sPsi + s * KTOT + k0 + DIM) = vi.v;
    }
  }
  __syncthreads();

  // ---- phase 3: WMMA GEMMs (double-buffered B) + weighted |.|^2 reduce ----
  {
    const int wv   = tid >> 5;     // 8 waves
    const int lane = tid & 31;
    const int tile = wv >> 1;      // 16-sample tile (0..3)
    const int nh   = wv & 1;       // which half of the 8 n-tiles
    const int lhi  = lane >> 4;
    const int llo  = lane & 15;
    const int arow = tile * 16 + llo;   // A layout: lanes 0-15/16-31 both M=llo

    // preload all 8 A fragments from LDS; live in VGPRs across all n-tiles
    AFrag afr[8];
#pragma unroll
    for (int ks = 0; ks < 8; ++ks) {
      const _Float16* ap = sPsi + arow * KTOT + ks * 32 + 16 * lhi;
#pragma unroll
      for (int h = 0; h < 8; ++h) { afr[ks].e[h] = ap[h]; afr[ks].e[h + 8] = ap[h + 8]; }
    }

    // hoist g_n = sum_w linW[w]*(1-2*bit_w(n)) for this wave's 4 n-tiles
    float garr[4];
#pragma unroll
    for (int q = 0; q < 4; ++q) {
      const int n = (nh * 4 + q) * 16 + llo;
      float g = 0.0f;
#pragma unroll
      for (int w = 0; w < NQ; ++w)
        g += linW[w] * (((n >> (NQ - 1 - w)) & 1) ? -1.0f : 1.0f);
      garr[q] = g;
    }

    float sacc[8];
#pragma unroll
    for (int r = 0; r < 8; ++r) sacc[r] = 0.0f;

    for (int q = 0; q < 4; ++q) {
      const int nt = nh * 4 + q;
      const int n  = nt * 16 + llo;
      const _Float16* browp  = Br + (size_t)n * KTOT + 16 * lhi;
      const _Float16* birowp = Bi + (size_t)n * KTOT + 16 * lhi;
      if (q < 3) {   // pull the NEXT n-tile's rows toward L2/L0 early
        __builtin_prefetch(browp  + 16 * KTOT, 0, 1);
        __builtin_prefetch(birowp + 16 * KTOT, 0, 1);
      }
      // double-buffered B fragments: loads for k-step ks+1 issue before the
      // WMMAs of k-step ks, so waits become partial instead of loadcnt==0.
      AFrag brf[2], bif[2];
#pragma unroll
      for (int h = 0; h < 16; ++h) { brf[0].e[h] = browp[h]; bif[0].e[h] = birowp[h]; }
      CFrag accR, accI;
#pragma unroll
      for (int r = 0; r < 8; ++r) { accR.e[r] = 0.0f; accI.e[r] = 0.0f; }
#pragma unroll
      for (int ks = 0; ks < 8; ++ks) {            // K = 256 in steps of 32
        const int cur = ks & 1;
        const int nxt = cur ^ 1;
        if (ks < 7) {
          const int kb = (ks + 1) * 32;
#pragma unroll
          for (int h = 0; h < 16; ++h) {
            brf[nxt].e[h] = browp[kb + h];
            bif[nxt].e[h] = birowp[kb + h];
          }
        }
        accR.v = __builtin_amdgcn_wmma_f32_16x16x32_f16(
            false, afr[ks].v, false, brf[cur].v, (short)0, accR.v, false, false);
        accI.v = __builtin_amdgcn_wmma_f32_16x16x32_f16(
            false, afr[ks].v, false, bif[cur].v, (short)0, accI.v, false, false);
      }
#pragma unroll
      for (int r = 0; r < 8; ++r) {
        const float prr = accR.e[r], pii = accI.e[r];
        sacc[r] += (prr * prr + pii * pii) * garr[q];
      }
    }
    // reduce over the 16 lanes of each half (C/D: lanes0-15 M=r, 16-31 M=8+r)
#pragma unroll
    for (int r = 0; r < 8; ++r) {
      sacc[r] += __shfl_xor(sacc[r], 1, 32);
      sacc[r] += __shfl_xor(sacc[r], 2, 32);
      sacc[r] += __shfl_xor(sacc[r], 4, 32);
      sacc[r] += __shfl_xor(sacc[r], 8, 32);
    }
    if (llo == 0) {
#pragma unroll
      for (int r = 0; r < 8; ++r) spart[wv][lhi * 8 + r] = sacc[r];
    }
  }
  __syncthreads();

  if (tid < SAMPB) {
    const int t = tid >> 4;
    const int m = tid & 15;
    const float v = spart[2 * t][m] + spart[2 * t + 1][m] + linb[0];
    __builtin_nontemporal_store(v, &out[base + tid]);
  }
}

// ---------------------------------------------------------------------------
extern "C" void kernel_launch(void* const* d_in, const int* in_sizes, int n_in,
                              void* d_out, int out_size, void* d_ws, size_t ws_size,
                              hipStream_t stream) {
  const float* x    = (const float*)d_in[0];
  const float* preW = (const float*)d_in[1];
  const float* preb = (const float*)d_in[2];
  const float* qw   = (const float*)d_in[3];
  const float* linW = (const float*)d_in[4];
  const float* linb = (const float*)d_in[5];
  float* out = (float*)d_out;

  _Float16* Br = (_Float16*)d_ws;                 // 128*256 f16 = 64KB
  _Float16* Bi = Br + (size_t)DIM * KTOT;         // 64KB more

  const int B = in_sizes[0] / D_IN;               // 65536

  hipLaunchKernelGGL(build_unitary_kernel, dim3(DIM), dim3(DIM), 0, stream,
                     qw, Br, Bi);
  hipLaunchKernelGGL(qnn_main_kernel, dim3(B / SAMPB), dim3(256), 0, stream,
                     x, preW, preb, linW, linb, Br, Bi, out);
}
